// SelfAttentionModule_65481071401534
// MI455X (gfx1250) — compile-verified, matrix operand order checked by reading, not run on hardware
//
#include <hip/hip_runtime.h>
#include <hip/hip_bf16.h>

// ---------------------------------------------------------------------------
// Multi-head self-attention for MI455X (gfx1250, wave32, WMMA bf16 path).
//   B=4, S=2048, D=1024, H=16, hd=64
// Pipeline: f32->bf16 convert -> Q/K/V WMMA GEMMs -> flash-attention with
// TDM (tensor_load_to_lds) double-buffered K/V tiles -> output projection
// WMMA GEMM (+residual, f32 out).
// ---------------------------------------------------------------------------

typedef __bf16 bf16_t;
typedef __attribute__((ext_vector_type(16))) __bf16 v16bf;
typedef __attribute__((ext_vector_type(8)))  float  v8f;
typedef __attribute__((ext_vector_type(4)))  unsigned int v4u;
typedef __attribute__((ext_vector_type(8)))  int  v8i;
typedef __attribute__((ext_vector_type(4)))  int  v4i;

#define DEV __device__ __forceinline__

constexpr int Dm  = 1024;   // embed dim
constexpr int Sm  = 2048;   // sequence
constexpr int Bm  = 4;      // batch
constexpr int Hm  = 16;     // heads
constexpr int HDm = 64;     // head dim
constexpr int Mrows = Bm * Sm;  // 8192

union FragU { v16bf v; uint4 q[2]; };

// Load a 16x32 bf16 WMMA A/B fragment.
// Element (r, k) lives at base + r*stride + k, contiguous along k.
// CDNA5 16-bit layout: lanes 0-15 hold r=lane, K 0..7 (v0-3) and 16..23
// (v4-7); lanes 16-31 hold r=lane-16, K 8..15 and 24..31.
DEV v16bf load_frag(const bf16_t* base, size_t stride, int lane) {
  const int r = lane & 15;
  const int k = (lane >> 4) << 3;
  const bf16_t* p = base + (size_t)r * stride + k;
  FragU f;
  f.q[0] = *(const uint4*)(p);
  f.q[1] = *(const uint4*)(p + 16);
  return f.v;
}

DEV v8f wmma_bf16(v16bf a, v16bf b, v8f c) {
  // (neg_a, A, neg_b, B, c_mod, C, reuse_a, reuse_b)
  return __builtin_amdgcn_wmma_f32_16x16x32_bf16(false, a, false, b,
                                                 (short)0, c, false, false);
}

// ---------------------------------------------------------------------------
// TDM: async 2-D tile load (bf16) global -> LDS, per CDNA5 ISA D# layout.
// group0: count=1 | lds_addr | global_addr[56:0] | type=2
// group1: data_size=1(2B), tensor_dim0=cols, tensor_dim1=rows,
//         tile_dim0=cols, tile_dim1=rows, tile_dim2=0, dim0_stride=stride.
// Tracked with TENSORcnt; EXEC-independent; issue from a uniform branch.
// Toolchain here is the 6-arg builtin (clang-23 / therock-10.0 headers).
// ---------------------------------------------------------------------------
DEV void tdm_load_2d_bf16(unsigned int lds_byte_off, const void* gptr,
                          unsigned int rows, unsigned int cols,
                          unsigned int stride_elems) {
  const unsigned long long ga = (unsigned long long)gptr;
  v4u g0;
  g0[0] = 1u;                                   // count=1, user mode
  g0[1] = lds_byte_off;                         // lds_addr
  g0[2] = (unsigned int)ga;                     // global_addr[31:0]
  g0[3] = (unsigned int)((ga >> 32) & 0x01FFFFFFull) | (2u << 30); // type=2
  v8i g1;
  g1[0] = 0x00010000;                           // data_size=1 (2 bytes)
  g1[1] = (int)(cols << 16);                    // tensor_dim0[15:0]
  g1[2] = (int)(rows << 16);                    // tensor_dim1[15:0]
  g1[3] = (int)(cols << 16);                    // tile_dim0
  g1[4] = (int)rows;                            // tile_dim1 (tile_dim2=0)
  g1[5] = (int)stride_elems;                    // tensor_dim0_stride[31:0]
  g1[6] = 0;                                    // stride hi / dim1_stride lo
  g1[7] = 0;
  v4i gz4 = (v4i)0;
  v8i gz8 = (v8i)0;
  __builtin_amdgcn_tensor_load_to_lds(g0, g1, gz4, gz4, gz8, 0);
}

// ---------------------------------------------------------------------------
// f32 -> bf16 convert (vectorized x4)
// ---------------------------------------------------------------------------
__global__ void cvt_bf16_kernel(const float* __restrict__ src,
                                bf16_t* __restrict__ dst, int n4) {
  int i = blockIdx.x * blockDim.x + threadIdx.x;
  const int stride = gridDim.x * blockDim.x;
  for (; i < n4; i += stride) {
    float4 v = ((const float4*)src)[i];
    union { bf16_t h[4]; uint2 u; } t;
    t.h[0] = (bf16_t)v.x; t.h[1] = (bf16_t)v.y;
    t.h[2] = (bf16_t)v.z; t.h[3] = (bf16_t)v.w;
    ((uint2*)dst)[i] = t.u;
  }
}

// ---------------------------------------------------------------------------
// GEMM: C[m,n] = sum_k A[m,k] * W[n,k]   (A: [8192,1024] bf16, W: [1024,1024])
// Block: 256 threads (8 waves), tile 128(M) x 128(N); wave tile 32x64.
// mode 0: store bf16 [m, n] scaled by `scale`           (Q / K)
// mode 1: store bf16 transposed per head: Vt[bh][d][s]  (V)
// mode 2: store f32 + residual to d_out                 (output projection)
// ---------------------------------------------------------------------------
__global__ __launch_bounds__(256) void gemm_wmma_kernel(
    const bf16_t* __restrict__ A, const bf16_t* __restrict__ W,
    void* __restrict__ outp, const float* __restrict__ resid,
    int mode, float scale) {
  const int lane = threadIdx.x & 31;
  const int wave = threadIdx.x >> 5;

  const int row0 = blockIdx.x * 128 + (wave >> 1) * 32;  // M base (2 frags)
  const int n0   = blockIdx.y * 128 + (wave & 1) * 64;   // N base (4 frags)

  v8f c[2][4];
#pragma unroll
  for (int i = 0; i < 2; ++i)
#pragma unroll
    for (int j = 0; j < 4; ++j) c[i][j] = (v8f)0.0f;

  for (int k0 = 0; k0 < Dm; k0 += 32) {
    v16bf a0 = load_frag(A + (size_t)row0 * Dm + k0, Dm, lane);
    v16bf a1 = load_frag(A + (size_t)(row0 + 16) * Dm + k0, Dm, lane);
    v16bf b[4];
#pragma unroll
    for (int j = 0; j < 4; ++j)
      b[j] = load_frag(W + (size_t)(n0 + j * 16) * Dm + k0, Dm, lane);
#pragma unroll
    for (int j = 0; j < 4; ++j) {
      c[0][j] = wmma_bf16(a0, b[j], c[0][j]);
      c[1][j] = wmma_bf16(a1, b[j], c[1][j]);
    }
  }

  // C layout: lanes 0-15 -> rows mb+0..7, lanes 16-31 -> rows mb+8..15
  const int n  = lane & 15;
  const int mb = (lane >> 4) << 3;

  if (mode == 0) {                       // bf16, natural [m, n] layout
    bf16_t* O = (bf16_t*)outp;
#pragma unroll
    for (int i = 0; i < 2; ++i)
#pragma unroll
      for (int j = 0; j < 4; ++j)
#pragma unroll
        for (int r = 0; r < 8; ++r) {
          int row = row0 + i * 16 + mb + r;
          O[(size_t)row * Dm + n0 + j * 16 + n] = (bf16_t)(c[i][j][r] * scale);
        }
  } else if (mode == 1) {                // Vt[bh][d][s] transposed store
    bf16_t* Vt = (bf16_t*)outp;
    const int bidx = row0 >> 11;         // batch (tile never crosses batch)
#pragma unroll
    for (int i = 0; i < 2; ++i) {
      const int srow = (row0 + i * 16 + mb) & (Sm - 1);
#pragma unroll
      for (int j = 0; j < 4; ++j) {
        const int e = n0 + j * 16 + n;   // global output column
        const int h = e >> 6, d = e & 63;
        union { bf16_t h8[8]; uint4 q; } t;
#pragma unroll
        for (int r = 0; r < 8; ++r) t.h8[r] = (bf16_t)c[i][j][r];
        *(uint4*)((bf16_t*)Vt +
                  ((size_t)(bidx * Hm + h) * HDm + d) * Sm + srow) = t.q;
      }
    }
  } else {                               // f32 + residual
    float* O = (float*)outp;
#pragma unroll
    for (int i = 0; i < 2; ++i)
#pragma unroll
      for (int j = 0; j < 4; ++j)
#pragma unroll
        for (int r = 0; r < 8; ++r) {
          size_t idx = (size_t)(row0 + i * 16 + mb + r) * Dm + n0 + j * 16 + n;
          O[idx] = c[i][j][r] + resid[idx];
        }
  }
}

// ---------------------------------------------------------------------------
// Flash attention: grid (S/64 q-tiles, B*H). 4 waves/block, each wave owns a
// 16-row Q strip. K/V 64x64 tiles staged to LDS by the TDM (double-buffered,
// issued by wave 0, TENSORcnt + workgroup barrier for sync); all 4 waves
// share the LDS tiles -> 4x less global traffic than direct loads.
// LDS map (bytes): K0 @0, K1 @8192, V0 @16384, V1 @24576, P @32768.
// ---------------------------------------------------------------------------
__global__ __launch_bounds__(128) void flash_attn_kernel(
    const bf16_t* __restrict__ Qb, const bf16_t* __restrict__ Kb,
    const bf16_t* __restrict__ Vt, bf16_t* __restrict__ ctx) {
  const int lane = threadIdx.x & 31;
  const int wave = threadIdx.x >> 5;
  const int qt = blockIdx.x;           // 0..31
  const int bh = blockIdx.y;           // 0..63
  const int b = bh >> 4, h = bh & 15;

  __shared__ __align__(16) bf16_t smem[4 * 4096 + 4 * 16 * 64];
  bf16_t* Plw = smem + 16384 + wave * (16 * 64);   // per-wave P staging

  const int n  = lane & 15;
  const int mb = (lane >> 4) << 3;
  const int q0 = qt * 64 + wave * 16;
  constexpr int KT = Sm / 64;

  const bf16_t* Kg = Kb + ((size_t)(b * Sm)) * Dm + h * HDm;      // [kpos, d]
  const bf16_t* Vg = Vt + ((size_t)bh * HDm) * Sm;                // [d, s]

  // Q strip in registers: 16 rows x 64 (hd) = two A fragments
  const bf16_t* Qbase = Qb + ((size_t)(b * Sm + q0)) * Dm + h * HDm;
  const v16bf qa0 = load_frag(Qbase + 0, Dm, lane);
  const v16bf qa1 = load_frag(Qbase + 32, Dm, lane);

  v8f o0 = (v8f)0.0f, o1 = (v8f)0.0f, o2 = (v8f)0.0f, o3 = (v8f)0.0f;
  float m[8], l[8];
#pragma unroll
  for (int r = 0; r < 8; ++r) { m[r] = -1e30f; l[r] = 0.0f; }

  // Prologue: stage tile 0 into buffer 0
  if (wave == 0) {
    tdm_load_2d_bf16(0u,     Kg, 64, 64, Dm);            // K tile [64 kpos][64 d]
    tdm_load_2d_bf16(16384u, Vg, 64, 64, Sm);            // V tile [64 d][64 s]
    __builtin_amdgcn_s_wait_tensorcnt(0);
  }
  __syncthreads();

  for (int kt = 0; kt < KT; ++kt) {
    const int cur = kt & 1;
    // Prefetch next tile into the other buffer while we compute this one.
    if (wave == 0 && (kt + 1) < KT) {
      const int nxt = (kt + 1) & 1;
      tdm_load_2d_bf16((unsigned)(nxt * 8192),
                       Kg + (size_t)(kt + 1) * 64 * Dm, 64, 64, Dm);
      tdm_load_2d_bf16((unsigned)(16384 + nxt * 8192),
                       Vg + (size_t)(kt + 1) * 64, 64, 64, Sm);
    }
    const bf16_t* ldsK = smem + cur * 4096;          // [kpos][64 d]
    const bf16_t* ldsV = smem + 8192 + cur * 4096;   // [d][64 s]

    // ---- scores: S(16x64) = Q(16x64) * K^T ---- (B frags from LDS)
    v8f s[4];
#pragma unroll
    for (int j = 0; j < 4; ++j) {
      v16bf bk0 = load_frag(ldsK + j * 16 * 64 + 0, 64, lane);
      v16bf bk1 = load_frag(ldsK + j * 16 * 64 + 32, 64, lane);
      v8f cc = (v8f)0.0f;
      cc = wmma_bf16(qa0, bk0, cc);
      cc = wmma_bf16(qa1, bk1, cc);
      s[j] = cc;
    }

    // ---- online softmax: row max, rescale, exp, row sum ----
    float sc[8];
#pragma unroll
    for (int r = 0; r < 8; ++r) {
      float mx = fmaxf(fmaxf(s[0][r], s[1][r]), fmaxf(s[2][r], s[3][r]));
#pragma unroll
      for (int d = 1; d < 16; d <<= 1) mx = fmaxf(mx, __shfl_xor(mx, d, 32));
      float mn = fmaxf(m[r], mx);
      sc[r] = __expf(m[r] - mn);
      m[r] = mn;
    }
#pragma unroll
    for (int r = 0; r < 8; ++r) {
      o0[r] *= sc[r]; o1[r] *= sc[r]; o2[r] *= sc[r]; o3[r] *= sc[r];
    }
#pragma unroll
    for (int r = 0; r < 8; ++r) {
      float rs = 0.0f;
#pragma unroll
      for (int j = 0; j < 4; ++j) {
        float p = __expf(s[j][r] - m[r]);
        Plw[(mb + r) * 64 + j * 16 + n] = (bf16_t)p;   // C-layout -> LDS
        rs += p;
      }
#pragma unroll
      for (int d = 1; d < 16; d <<= 1) rs += __shfl_xor(rs, d, 32);
      l[r] = l[r] * sc[r] + rs;
    }

    // ---- P (A-layout, via LDS transpose) x V ----
    v16bf pa0 = load_frag(Plw + 0, 64, lane);
    v16bf pa1 = load_frag(Plw + 32, 64, lane);
#pragma unroll
    for (int t = 0; t < 4; ++t) {
      v16bf bv0 = load_frag(ldsV + t * 16 * 64 + 0, 64, lane);
      v16bf bv1 = load_frag(ldsV + t * 16 * 64 + 32, 64, lane);
      v8f* op = (t == 0) ? &o0 : (t == 1) ? &o1 : (t == 2) ? &o2 : &o3;
      *op = wmma_bf16(pa0, bv0, *op);
      *op = wmma_bf16(pa1, bv1, *op);
    }

    // Next tile must be resident, and everyone must be done reading `cur`
    // before iteration kt+1 overwrites buffer cur^1's partner.
    if (wave == 0) __builtin_amdgcn_s_wait_tensorcnt(0);
    __syncthreads();
  }

  // ---- finalize: O /= l, store bf16 context [b, q, h*64 + d] ----
  bf16_t* cb = ctx + ((size_t)(b * Sm + q0)) * Dm + h * HDm;
#pragma unroll
  for (int r = 0; r < 8; ++r) {
    const float inv = 1.0f / l[r];
    const size_t rowoff = (size_t)(mb + r) * Dm;
    cb[rowoff + 0 * 16 + n] = (bf16_t)(o0[r] * inv);
    cb[rowoff + 1 * 16 + n] = (bf16_t)(o1[r] * inv);
    cb[rowoff + 2 * 16 + n] = (bf16_t)(o2[r] * inv);
    cb[rowoff + 3 * 16 + n] = (bf16_t)(o3[r] * inv);
  }
}

// ---------------------------------------------------------------------------
// Launch
// ---------------------------------------------------------------------------
extern "C" void kernel_launch(void* const* d_in, const int* in_sizes, int n_in,
                              void* d_out, int out_size, void* d_ws,
                              size_t ws_size, hipStream_t stream) {
  (void)in_sizes; (void)n_in; (void)out_size; (void)ws_size;
  const float* x  = (const float*)d_in[2];
  const float* Wq = (const float*)d_in[3];
  const float* Wk = (const float*)d_in[4];
  const float* Wv = (const float*)d_in[5];
  const float* Wo = (const float*)d_in[6];

  char* ws = (char*)d_ws;
  const size_t XB  = (size_t)Mrows * Dm * sizeof(bf16_t);   // 16 MB
  const size_t WB  = (size_t)Dm * Dm * sizeof(bf16_t);      // 2 MB
  bf16_t* xb  = (bf16_t*)(ws);
  bf16_t* wqb = (bf16_t*)(ws + XB);
  bf16_t* wkb = (bf16_t*)(ws + XB + WB);
  bf16_t* wvb = (bf16_t*)(ws + XB + 2 * WB);
  bf16_t* wob = (bf16_t*)(ws + XB + 3 * WB);
  bf16_t* Qb  = (bf16_t*)(ws + XB + 4 * WB);
  bf16_t* Kb  = (bf16_t*)(ws + 2 * XB + 4 * WB);
  bf16_t* Vt  = (bf16_t*)(ws + 3 * XB + 4 * WB);
  bf16_t* ctx = (bf16_t*)(ws + 4 * XB + 4 * WB);

  // f32 -> bf16 conversions
  cvt_bf16_kernel<<<1024, 256, 0, stream>>>(x,  xb,  Mrows * Dm / 4);
  cvt_bf16_kernel<<<256, 256, 0, stream>>>(Wq, wqb, Dm * Dm / 4);
  cvt_bf16_kernel<<<256, 256, 0, stream>>>(Wk, wkb, Dm * Dm / 4);
  cvt_bf16_kernel<<<256, 256, 0, stream>>>(Wv, wvb, Dm * Dm / 4);
  cvt_bf16_kernel<<<256, 256, 0, stream>>>(Wo, wob, Dm * Dm / 4);

  const dim3 ggrid(Mrows / 128, Dm / 128);   // (64, 8)
  // Q scaled by 1/sqrt(hd); K plain; V transposed per head
  gemm_wmma_kernel<<<ggrid, 256, 0, stream>>>(xb, wqb, Qb, nullptr, 0, 0.125f);
  gemm_wmma_kernel<<<ggrid, 256, 0, stream>>>(xb, wkb, Kb, nullptr, 0, 1.0f);
  gemm_wmma_kernel<<<ggrid, 256, 0, stream>>>(xb, wvb, Vt, nullptr, 1, 1.0f);

  flash_attn_kernel<<<dim3(Sm / 64, Bm * Hm), 128, 0, stream>>>(Qb, Kb, Vt, ctx);

  // output projection + residual, f32 out
  gemm_wmma_kernel<<<ggrid, 256, 0, stream>>>(ctx, wob, d_out, x, 2, 1.0f);
}
